// ALEncoder_27908697489983
// MI455X (gfx1250) — compile-verified, matrix operand order checked by reading, not run on hardware
//
#include <hip/hip_runtime.h>
#include <hip/hip_bf16.h>
#include <math.h>

typedef __attribute__((ext_vector_type(16))) _Float16 v16h;
typedef __attribute__((ext_vector_type(4)))  _Float16 v4h;
typedef __attribute__((ext_vector_type(2)))  _Float16 v2h;
typedef __attribute__((ext_vector_type(8)))  float    v8f;

#define N_AG 64
#define L_LN 4096
#define D_MD 128
#define R2   900.0f

// LDS layout constants
#define WBS    130                 // v2h per k-pair row (padded: 128+2)
#define WB_ONE (64 * WBS)          // v2h elems per weight matrix
#define NW     5                   // mie_w2[0], mie_w2[1], aggr_w, wk, wv
#define HS     132                 // hbuf f32 row stride (padded)
#define AS     132                 // a16 _Float16 row stride (padded, even)
#define PWB    (16*HS*4 + 16*(AS/2)*4 + 64)   // per-wave scratch bytes = 12736
#define WB_BYTES (NW * WB_ONE * 4)            // 166400
#define RED_BYTES (1160 * 4)                  // m[64]+s[64]+cnt[8]+o[1024]
#define PAR_FLOATS 2048
#define PAR_BYTES (PAR_FLOATS * 4)            // 8192
#define SMEM_BYTES (WB_BYTES + 8*PWB + RED_BYTES + PAR_BYTES)  // 281120

// ---- A-tile loader: 16x32 f16 A-layout (ISA 7.12.2), A from LDS row-major ----
__device__ __forceinline__ v16h load_a_tile(const v2h* a16, int kt, int lane) {
    const int r = lane & 15, hl = lane >> 4;
    v16h A;
#pragma unroll
    for (int i = 0; i < 8; ++i) {
        int kl = ((i & 3) << 1) + ((i >> 2) << 4) + (hl << 3);
        int k  = (kt << 5) + kl;
        v2h p = a16[r * (AS / 2) + (k >> 1)];
        A[2 * i] = p[0]; A[2 * i + 1] = p[1];
    }
    return A;
}

// ---- B-tile loader: 32x16 f16 B-layout, weights pre-transposed [k][n] in LDS ----
__device__ __forceinline__ v16h load_b_tile(const v2h* wb, int wslot, int kt, int nt, int lane) {
    const int col = lane & 15, hl = lane >> 4;
    v16h B;
#pragma unroll
    for (int i = 0; i < 8; ++i) {
        int kl = ((i & 3) << 1) + ((i >> 2) << 4) + (hl << 3);
        int k  = (kt << 5) + kl;
        v2h p = wb[wslot * WB_ONE + (k >> 1) * WBS + (nt << 4) + col];
        B[2 * i] = p[0]; B[2 * i + 1] = p[1];
    }
    return B;
}

// ======================= prep kernels =======================
__global__ __launch_bounds__(256) void prep_cat_kernel(
    const float* __restrict__ ei, const float* __restrict__ et, const float* __restrict__ ec,
    const int* __restrict__ ii, const int* __restrict__ td, const int* __restrict__ tc,
    float* __restrict__ cat_ws)
{
    int idx = blockIdx.x * blockDim.x + threadIdx.x;
    if (idx >= L_LN * D_MD) return;
    int l = idx >> 7, d = idx & 127;
    cat_ws[idx] = ei[ii[l] * D_MD + d] + et[td[l] * D_MD + d] + ec[tc[l] * D_MD + d];
}

__global__ __launch_bounds__(128) void prep_q_kernel(
    const float* __restrict__ temporal, const float* __restrict__ n1_g,
    const float* __restrict__ n1_b, const float* __restrict__ wq, float* __restrict__ q_ws)
{
    __shared__ float sx[128], sxn[128];
    int n = blockIdx.x, d = threadIdx.x;
    sx[d] = temporal[n * 128 + d];
    __syncthreads();
    float ps = 0.f, ps2 = 0.f;
    for (int k = 0; k < 128; ++k) { float v = sx[k]; ps += v; ps2 += v * v; }
    float mean = ps * 0.0078125f;
    float inv  = rsqrtf(ps2 * 0.0078125f - mean * mean + 1e-5f);
    sxn[d] = (sx[d] - mean) * inv * n1_g[d] + n1_b[d];
    __syncthreads();
    float acc = 0.f;
    for (int k = 0; k < 128; ++k) acc += sxn[k] * wq[d * 128 + k];
    q_ws[n * 128 + d] = acc;
}

// ======================= main fused kernel =======================
__global__ __launch_bounds__(256) void al_main_kernel(
    const float* __restrict__ positions, const float* __restrict__ lane_start,
    const float* __restrict__ lane_end,
    const float* __restrict__ mie_w1, const float* __restrict__ mie_b1,
    const float* __restrict__ mie_g1, const float* __restrict__ mie_be1,
    const float* __restrict__ mie_w2, const float* __restrict__ mie_b2,
    const float* __restrict__ aggr_g1, const float* __restrict__ aggr_be1,
    const float* __restrict__ aggr_w, const float* __restrict__ aggr_b,
    const float* __restrict__ aggr_g2, const float* __restrict__ aggr_be2,
    const float* __restrict__ wk, const float* __restrict__ wv,
    const float* __restrict__ cat_ws, const float* __restrict__ q_ws,
    float* __restrict__ o_ws)
{
    extern __shared__ __align__(16) char smem[];
    const int tid  = threadIdx.x;
    const int lane = tid & 31;
    const int w    = tid >> 5;
    const int r    = lane & 15;
    const int hl   = lane >> 4;
    const int col  = r;
    const int n    = blockIdx.x;

    v2h*  wb     = (v2h*)smem;
    char* pwbase = smem + (size_t)WB_BYTES;
    char* mybase = pwbase + (size_t)w * PWB;
    float*     hbuf    = (float*)mybase;
    v2h*       a16     = (v2h*)(mybase + 16 * HS * 4);
    _Float16*  a16h    = (_Float16*)a16;
    int*       maskbuf = (int*)(mybase + 16 * HS * 4 + 16 * (AS / 2) * 4);
    float* red   = (float*)(pwbase + 8 * PWB);
    float* red_m = red;
    float* red_s = red + 64;
    float* red_c = red + 128;
    float* red_o = red + 136;
    float* pbuf  = (float*)(pwbase + 8 * PWB + RED_BYTES);

    // ---- one-time weight convert f32 -> f16 (transposed [k][n]) into LDS ----
    const float* wptr[NW] = { mie_w2, mie_w2 + D_MD * D_MD, aggr_w, wk, wv };
    for (int idx = tid; idx < NW * 64 * 128; idx += 256) {
        int ws  = idx >> 13;
        int rem = idx & 8191;
        int kp  = rem >> 7;
        int nn  = rem & 127;
        const float* W = wptr[ws];
        __builtin_prefetch(&W[nn * 128 + 2 * kp + 512], 0, 1);
        v2h p;
        p[0] = (_Float16)W[nn * 128 + 2 * kp];
        p[1] = (_Float16)W[nn * 128 + 2 * kp + 1];
        wb[ws * WB_ONE + kp * WBS + nn] = p;
    }
    // ---- one-time per-feature parameter staging into LDS ----
    if (tid < 128) {
        int d = tid;
#pragma unroll
        for (int bi = 0; bi < 2; ++bi) {
            float* pb = pbuf + bi * 640;
            pb[d]       = mie_w1[bi * 256 + d * 2];
            pb[128 + d] = mie_w1[bi * 256 + d * 2 + 1];
            pb[256 + d] = mie_b1[bi * 128 + d];
            pb[384 + d] = mie_g1[bi * 128 + d];
            pb[512 + d] = mie_be1[bi * 128 + d];
        }
        pbuf[1280 + d] = aggr_g1[d];
        pbuf[1408 + d] = aggr_be1[d];
        pbuf[1536 + d] = aggr_g2[d];
        pbuf[1664 + d] = aggr_be2[d];
        pbuf[1792 + d] = aggr_b[d];
        pbuf[1920 + d] = mie_b2[d] + mie_b2[128 + d];
    }

    // ---- per-agent rotation + query ----
    float p0x = positions[n * 4 + 0], p0y = positions[n * 4 + 1];
    float px  = positions[n * 4 + 2], py  = positions[n * 4 + 3];
    float hx = px - p0x, hy = py - p0y;
    float hr2 = hx * hx + hy * hy;
    float rn  = (hr2 > 1e-20f) ? rsqrtf(hr2) : 0.0f;
    float cth = (hr2 > 1e-20f) ? hx * rn : 1.0f;
    float sth = (hr2 > 1e-20f) ? hy * rn : 0.0f;

    float qreg[8];
#pragma unroll
    for (int h = 0; h < 8; ++h) qreg[h] = q_ws[n * 128 + h * 16 + col];

    float m_h[8], s_h[8], oacc[8];
#pragma unroll
    for (int h = 0; h < 8; ++h) { m_h[h] = -1e30f; s_h[h] = 0.f; oacc[h] = 0.f; }
    float cntf = 0.f;

    __syncthreads();   // wb + pbuf visible to all waves

    // ---- tile loop: wave w handles tiles w, w+8, ... (32 each), fully decoupled ----
    for (int it = 0; it < (L_LN / 16) / 8; ++it) {
        int tile = it * 8 + w;
        int l0 = tile * 16;
        int l = l0 + r;

        float ls0 = lane_start[2 * l], ls1 = lane_start[2 * l + 1];
        float le0 = lane_end[2 * l],   le1 = lane_end[2 * l + 1];
        float dx = le0 - ls0, dy = le1 - ls1;
        float vv0 =  dx * cth + dy * sth, vv1 = -dx * sth + dy * cth;
        float rx = ls0 - px, ry = ls1 - py;
        float rr0 =  rx * cth + ry * sth, rr1 = -rx * sth + ry * cth;
        int msk = (rx * rx + ry * ry) < R2 ? 1 : 0;
        if (hl == 0) maskbuf[r] = msk;
        unsigned long long mb = __ballot(msk != 0);
        cntf += (float)__popcll(mb & 0xFFFFull);

        float bx[2] = { vv0, rr0 }, by[2] = { vv1, rr1 };

        // ---- two MIE branches: Linear(2->128) (store raw f16) + LN in place + WMMA x mie_w2 ----
        for (int bi = 0; bi < 2; ++bi) {
            const float* pb = pbuf + bi * 640;
            float ps = 0.f, ps2 = 0.f;
            for (int dd = 0; dd < 64; dd += 2) {
                int d = hl * 64 + dd;
                float h0a = bx[bi] * pb[d]     + by[bi] * pb[128 + d]     + pb[256 + d];
                float h0b = bx[bi] * pb[d + 1] + by[bi] * pb[128 + d + 1] + pb[256 + d + 1];
                ps += h0a + h0b; ps2 += h0a * h0a + h0b * h0b;
                v2h o; o[0] = (_Float16)h0a; o[1] = (_Float16)h0b;
                a16[(r * AS + d) >> 1] = o;
            }
            ps  += __shfl_xor(ps, 16);
            ps2 += __shfl_xor(ps2, 16);
            float mean = ps * 0.0078125f;
            float inv  = rsqrtf(ps2 * 0.0078125f - mean * mean + 1e-5f);
            __builtin_amdgcn_wave_barrier();
            for (int dd = 0; dd < 64; dd += 4) {
                int d = hl * 64 + dd;
                float4 g  = *(const float4*)(pb + 384 + d);
                float4 be = *(const float4*)(pb + 512 + d);
                v4h pr = *(const v4h*)(a16h + r * AS + d);
                v4h o;
                o[0] = (_Float16)fmaxf(((float)pr[0] - mean) * inv * g.x + be.x, 0.f);
                o[1] = (_Float16)fmaxf(((float)pr[1] - mean) * inv * g.y + be.y, 0.f);
                o[2] = (_Float16)fmaxf(((float)pr[2] - mean) * inv * g.z + be.z, 0.f);
                o[3] = (_Float16)fmaxf(((float)pr[3] - mean) * inv * g.w + be.w, 0.f);
                *(v4h*)(a16h + r * AS + d) = o;
            }
            __builtin_amdgcn_wave_barrier();
            for (int nt = 0; nt < 8; ++nt) {
                v8f c;
                if (bi == 0) {
#pragma unroll
                    for (int j = 0; j < 8; ++j) c[j] = 0.f;
                } else {
#pragma unroll
                    for (int j = 0; j < 8; ++j) c[j] = hbuf[(j + hl * 8) * HS + nt * 16 + col];
                }
#pragma unroll
                for (int kt = 0; kt < 4; ++kt) {
                    v16h A = load_a_tile(a16, kt, lane);
                    v16h B = load_b_tile(wb, bi, kt, nt, lane);
                    c = __builtin_amdgcn_wmma_f32_16x16x32_f16(false, A, false, B, (short)0, c, false, false);
                }
                if (bi == 1) {
#pragma unroll
                    for (int j = 0; j < 8; ++j) {
                        int dcol = nt * 16 + col;
                        int lr = l0 + j + hl * 8;
                        c[j] += pbuf[1920 + dcol] + cat_ws[lr * 128 + dcol];
                    }
                }
#pragma unroll
                for (int j = 0; j < 8; ++j) hbuf[(j + hl * 8) * HS + nt * 16 + col] = c[j];
            }
            __builtin_amdgcn_wave_barrier();
        }

        // ---- h -> LN(aggr_g1) -> ReLU -> a16 (vectorized) ----
        {
            const float* hrow = hbuf + r * HS + hl * 64;
            float ps = 0.f, ps2 = 0.f;
#pragma unroll
            for (int q4 = 0; q4 < 16; ++q4) {
                float4 v = ((const float4*)hrow)[q4];
                ps  += v.x + v.y + v.z + v.w;
                ps2 += v.x * v.x + v.y * v.y + v.z * v.z + v.w * v.w;
            }
            ps += __shfl_xor(ps, 16); ps2 += __shfl_xor(ps2, 16);
            float mean = ps * 0.0078125f;
            float inv  = rsqrtf(ps2 * 0.0078125f - mean * mean + 1e-5f);
#pragma unroll
            for (int q4 = 0; q4 < 16; ++q4) {
                int d = hl * 64 + q4 * 4;
                float4 v  = ((const float4*)hrow)[q4];
                float4 g  = *(const float4*)(pbuf + 1280 + d);
                float4 be = *(const float4*)(pbuf + 1408 + d);
                v4h o;
                o[0] = (_Float16)fmaxf((v.x - mean) * inv * g.x + be.x, 0.f);
                o[1] = (_Float16)fmaxf((v.y - mean) * inv * g.y + be.y, 0.f);
                o[2] = (_Float16)fmaxf((v.z - mean) * inv * g.z + be.z, 0.f);
                o[3] = (_Float16)fmaxf((v.w - mean) * inv * g.w + be.w, 0.f);
                *(v4h*)(a16h + r * AS + d) = o;
            }
        }
        __builtin_amdgcn_wave_barrier();

        // ---- aggr matmul (WMMA) + bias -> hbuf ----
        for (int nt = 0; nt < 8; ++nt) {
            v8f c;
#pragma unroll
            for (int j = 0; j < 8; ++j) c[j] = 0.f;
#pragma unroll
            for (int kt = 0; kt < 4; ++kt) {
                v16h A = load_a_tile(a16, kt, lane);
                v16h B = load_b_tile(wb, 2, kt, nt, lane);
                c = __builtin_amdgcn_wmma_f32_16x16x32_f16(false, A, false, B, (short)0, c, false, false);
            }
            float bb = pbuf[1792 + nt * 16 + col];
#pragma unroll
            for (int j = 0; j < 8; ++j)
                hbuf[(j + hl * 8) * HS + nt * 16 + col] = c[j] + bb;
        }
        __builtin_amdgcn_wave_barrier();

        // ---- lane_feat = LN(aggr_g2) (no relu) -> a16 (vectorized) ----
        {
            const float* hrow = hbuf + r * HS + hl * 64;
            float ps = 0.f, ps2 = 0.f;
#pragma unroll
            for (int q4 = 0; q4 < 16; ++q4) {
                float4 v = ((const float4*)hrow)[q4];
                ps  += v.x + v.y + v.z + v.w;
                ps2 += v.x * v.x + v.y * v.y + v.z * v.z + v.w * v.w;
            }
            ps += __shfl_xor(ps, 16); ps2 += __shfl_xor(ps2, 16);
            float mean = ps * 0.0078125f;
            float inv  = rsqrtf(ps2 * 0.0078125f - mean * mean + 1e-5f);
#pragma unroll
            for (int q4 = 0; q4 < 16; ++q4) {
                int d = hl * 64 + q4 * 4;
                float4 v  = ((const float4*)hrow)[q4];
                float4 g  = *(const float4*)(pbuf + 1536 + d);
                float4 be = *(const float4*)(pbuf + 1664 + d);
                v4h o;
                o[0] = (_Float16)((v.x - mean) * inv * g.x + be.x);
                o[1] = (_Float16)((v.y - mean) * inv * g.y + be.y);
                o[2] = (_Float16)((v.z - mean) * inv * g.z + be.z);
                o[3] = (_Float16)((v.w - mean) * inv * g.w + be.w);
                *(v4h*)(a16h + r * AS + d) = o;
            }
        }
        __builtin_amdgcn_wave_barrier();

        // ---- attention: per head, K tile (WMMA) -> scores -> online softmax -> V tile (WMMA) ----
        v16h A0 = load_a_tile(a16, 0, lane);
        v16h A1 = load_a_tile(a16, 1, lane);
        v16h A2 = load_a_tile(a16, 2, lane);
        v16h A3 = load_a_tile(a16, 3, lane);
        for (int h = 0; h < 8; ++h) {
            v8f ck;
#pragma unroll
            for (int j = 0; j < 8; ++j) ck[j] = 0.f;
            ck = __builtin_amdgcn_wmma_f32_16x16x32_f16(false, A0, false, load_b_tile(wb, 3, 0, h, lane), (short)0, ck, false, false);
            ck = __builtin_amdgcn_wmma_f32_16x16x32_f16(false, A1, false, load_b_tile(wb, 3, 1, h, lane), (short)0, ck, false, false);
            ck = __builtin_amdgcn_wmma_f32_16x16x32_f16(false, A2, false, load_b_tile(wb, 3, 2, h, lane), (short)0, ck, false, false);
            ck = __builtin_amdgcn_wmma_f32_16x16x32_f16(false, A3, false, load_b_tile(wb, 3, 3, h, lane), (short)0, ck, false, false);

            float sc[8];
#pragma unroll
            for (int j = 0; j < 8; ++j) sc[j] = ck[j] * qreg[h];
#pragma unroll
            for (int st = 1; st < 16; st <<= 1) {
#pragma unroll
                for (int j = 0; j < 8; ++j) sc[j] += __shfl_xor(sc[j], st);
            }
            float tm = -1e30f;
#pragma unroll
            for (int j = 0; j < 8; ++j) {
                int rm = maskbuf[j + hl * 8];
                sc[j] = rm ? sc[j] * 0.25f : -1e9f;
                tm = fmaxf(tm, sc[j]);
            }
            tm = fmaxf(tm, __shfl_xor(tm, 16));
            float nm = fmaxf(m_h[h], tm);
            float scale = __expf(m_h[h] - nm);
            float p[8]; float tsum = 0.f;
#pragma unroll
            for (int j = 0; j < 8; ++j) { p[j] = __expf(sc[j] - nm); tsum += p[j]; }
            tsum += __shfl_xor(tsum, 16);
            s_h[h] = s_h[h] * scale + tsum;
            oacc[h] *= scale;
            m_h[h] = nm;

            v8f cv;
#pragma unroll
            for (int j = 0; j < 8; ++j) cv[j] = 0.f;
            cv = __builtin_amdgcn_wmma_f32_16x16x32_f16(false, A0, false, load_b_tile(wb, 4, 0, h, lane), (short)0, cv, false, false);
            cv = __builtin_amdgcn_wmma_f32_16x16x32_f16(false, A1, false, load_b_tile(wb, 4, 1, h, lane), (short)0, cv, false, false);
            cv = __builtin_amdgcn_wmma_f32_16x16x32_f16(false, A2, false, load_b_tile(wb, 4, 2, h, lane), (short)0, cv, false, false);
            cv = __builtin_amdgcn_wmma_f32_16x16x32_f16(false, A3, false, load_b_tile(wb, 4, 3, h, lane), (short)0, cv, false, false);
#pragma unroll
            for (int j = 0; j < 8; ++j) oacc[h] += p[j] * cv[j];
        }
        __builtin_amdgcn_wave_barrier();
    }

    // ---- cross-wave combine (flash-softmax merge) ----
#pragma unroll
    for (int h = 0; h < 8; ++h) oacc[h] += __shfl_xor(oacc[h], 16);
    if (lane < 16) {
#pragma unroll
        for (int h = 0; h < 8; ++h) red_o[(w * 8 + h) * 16 + lane] = oacc[h];
    }
    if (lane == 0) {
#pragma unroll
        for (int h = 0; h < 8; ++h) { red_m[w * 8 + h] = m_h[h]; red_s[w * 8 + h] = s_h[h]; }
        red_c[w] = cntf;
    }
    __syncthreads();
    if (tid < 128) {
        int hh = tid >> 4, cc = tid & 15;
        float M = -3e38f;
        for (int ww = 0; ww < 8; ++ww) M = fmaxf(M, red_m[ww * 8 + hh]);
        float S = 0.f, O = 0.f, CT = 0.f;
        for (int ww = 0; ww < 8; ++ww) {
            float e = __expf(red_m[ww * 8 + hh] - M);
            S += red_s[ww * 8 + hh] * e;
            O += red_o[(ww * 8 + hh) * 16 + cc] * e;
            CT += red_c[ww];
        }
        o_ws[n * 128 + hh * 16 + cc] = (CT > 0.5f && S > 0.f) ? (O / S) : 0.f;
    }
}

// ======================= final per-agent fusion + FFN =======================
__global__ __launch_bounds__(128) void al_final_kernel(
    const float* __restrict__ temporal, const float* __restrict__ o_ws,
    const float* __restrict__ wo, const float* __restrict__ w_self, const float* __restrict__ b_self,
    const float* __restrict__ w_ih, const float* __restrict__ b_ih,
    const float* __restrict__ w_hh, const float* __restrict__ b_hh,
    const float* __restrict__ n1_g, const float* __restrict__ n1_b,
    const float* __restrict__ n2_g, const float* __restrict__ n2_b,
    const float* __restrict__ mlp_w1, const float* __restrict__ mlp_b1,
    const float* __restrict__ mlp_w2, const float* __restrict__ mlp_b2,
    float* __restrict__ out)
{
    __shared__ float sx[128], sxn[128], so[128], soo[128], scen[128], scn[128], shid[512];
    int n = blockIdx.x, d = threadIdx.x;
    sx[d] = temporal[n * 128 + d];
    so[d] = o_ws[n * 128 + d];
    __syncthreads();
    float ps = 0.f, ps2 = 0.f;
    for (int k = 0; k < 128; ++k) { float v = sx[k]; ps += v; ps2 += v * v; }
    float mean = ps * 0.0078125f;
    float inv  = rsqrtf(ps2 * 0.0078125f - mean * mean + 1e-5f);
    sxn[d] = (sx[d] - mean) * inv * n1_g[d] + n1_b[d];
    __syncthreads();
    float oo = 0.f;
    for (int k = 0; k < 128; ++k) oo += so[k] * wo[d * 128 + k];
    soo[d] = oo;
    __syncthreads();
    float gi = b_ih[d], gh = b_hh[d], sf = b_self[d];
    for (int k = 0; k < 128; ++k) {
        gi += soo[k] * w_ih[d * 128 + k];
        gh += sxn[k] * w_hh[d * 128 + k];
        sf += sxn[k] * w_self[d * 128 + k];
    }
    float gate = 1.f / (1.f + __expf(-(gi + gh)));
    float mha = soo[d] + gate * (sf - soo[d]);
    float center = sx[d] + mha;
    scen[d] = center;
    __syncthreads();
    ps = 0.f; ps2 = 0.f;
    for (int k = 0; k < 128; ++k) { float v = scen[k]; ps += v; ps2 += v * v; }
    mean = ps * 0.0078125f;
    inv  = rsqrtf(ps2 * 0.0078125f - mean * mean + 1e-5f);
    scn[d] = (scen[d] - mean) * inv * n2_g[d] + n2_b[d];
    __syncthreads();
    for (int jj = 0; jj < 4; ++jj) {
        int j = d + jj * 128;
        float a = mlp_b1[j];
        for (int k = 0; k < 128; ++k) a += scn[k] * mlp_w1[j * 128 + k];
        shid[j] = fmaxf(a, 0.f);
    }
    __syncthreads();
    float y = mlp_b2[d];
    for (int j = 0; j < 512; ++j) y += shid[j] * mlp_w2[d * 512 + j];
    out[n * 128 + d] = center + y;
}

// ======================= launcher =======================
extern "C" void kernel_launch(void* const* d_in, const int* in_sizes, int n_in,
                              void* d_out, int out_size, void* d_ws, size_t ws_size,
                              hipStream_t stream) {
    (void)in_sizes; (void)n_in; (void)out_size; (void)ws_size;
    const float* temporal  = (const float*)d_in[0];
    const float* positions = (const float*)d_in[1];
    const float* lane_start = (const float*)d_in[2];
    const float* lane_end   = (const float*)d_in[3];
    const float* mie_w1  = (const float*)d_in[4];
    const float* mie_b1  = (const float*)d_in[5];
    const float* mie_g1  = (const float*)d_in[6];
    const float* mie_be1 = (const float*)d_in[7];
    const float* mie_w2  = (const float*)d_in[8];
    const float* mie_b2  = (const float*)d_in[9];
    const float* aggr_g1  = (const float*)d_in[10];
    const float* aggr_be1 = (const float*)d_in[11];
    const float* aggr_w   = (const float*)d_in[12];
    const float* aggr_b   = (const float*)d_in[13];
    const float* aggr_g2  = (const float*)d_in[14];
    const float* aggr_be2 = (const float*)d_in[15];
    const float* emb_int  = (const float*)d_in[16];
    const float* emb_turn = (const float*)d_in[17];
    const float* emb_tc   = (const float*)d_in[18];
    const float* wq = (const float*)d_in[19];
    const float* wk = (const float*)d_in[20];
    const float* wv = (const float*)d_in[21];
    const float* wo = (const float*)d_in[22];
    const float* w_self = (const float*)d_in[23];
    const float* b_self = (const float*)d_in[24];
    const float* w_ih = (const float*)d_in[25];
    const float* b_ih = (const float*)d_in[26];
    const float* w_hh = (const float*)d_in[27];
    const float* b_hh = (const float*)d_in[28];
    const float* n1_g = (const float*)d_in[29];
    const float* n1_b = (const float*)d_in[30];
    const float* n2_g = (const float*)d_in[31];
    const float* n2_b = (const float*)d_in[32];
    const float* mlp_w1 = (const float*)d_in[33];
    const float* mlp_b1 = (const float*)d_in[34];
    const float* mlp_w2 = (const float*)d_in[35];
    const float* mlp_b2 = (const float*)d_in[36];
    const int* is_int = (const int*)d_in[37];
    const int* turn_d = (const int*)d_in[38];
    const int* traf_c = (const int*)d_in[39];

    float* cat_ws = (float*)d_ws;                  // L*128 f32
    float* q_ws   = cat_ws + L_LN * D_MD;          // N*128 f32
    float* o_ws   = q_ws + N_AG * D_MD;            // N*128 f32

    prep_cat_kernel<<<(L_LN * D_MD + 255) / 256, 256, 0, stream>>>(
        emb_int, emb_turn, emb_tc, is_int, turn_d, traf_c, cat_ws);
    prep_q_kernel<<<N_AG, 128, 0, stream>>>(temporal, n1_g, n1_b, wq, q_ws);

    hipFuncSetAttribute(reinterpret_cast<const void*>(al_main_kernel),
                        hipFuncAttributeMaxDynamicSharedMemorySize, SMEM_BYTES);
    al_main_kernel<<<N_AG, 256, SMEM_BYTES, stream>>>(
        positions, lane_start, lane_end,
        mie_w1, mie_b1, mie_g1, mie_be1, mie_w2, mie_b2,
        aggr_g1, aggr_be1, aggr_w, aggr_b, aggr_g2, aggr_be2,
        wk, wv, cat_ws, q_ws, o_ws);

    al_final_kernel<<<N_AG, 128, 0, stream>>>(
        temporal, o_ws, wo, w_self, b_self, w_ih, b_ih, w_hh, b_hh,
        n1_g, n1_b, n2_g, n2_b, mlp_w1, mlp_b1, mlp_w2, mlp_b2, (float*)d_out);
}
